// MultiHeadSelfAttention_22050362097689
// MI455X (gfx1250) — matrix operand order repaired and checked by reading, not compiled, not run on hardware
//
#include <hip/hip_runtime.h>
#include <hip/hip_bf16.h>

typedef __attribute__((ext_vector_type(16))) _Float16 v16h;
typedef __attribute__((ext_vector_type(8)))  _Float16 v8h;
typedef __attribute__((ext_vector_type(4)))  _Float16 v4h;
typedef __attribute__((ext_vector_type(8)))  float    v8f;
typedef __attribute__((ext_vector_type(4)))  float    v4f;
typedef __attribute__((ext_vector_type(4)))  unsigned int u32x4;
typedef __attribute__((ext_vector_type(8)))  int      i32x8;
typedef __attribute__((ext_vector_type(4)))  int      i32x4;

#define NUM_HEADS 12
#define DEPTH     64
#define DMODEL    768
#define SEQ       2048
#define BATCH     2
#define ROWS      (BATCH*SEQ)          // 4096
#define BH        (BATCH*NUM_HEADS)    // 24

#if defined(__has_builtin)
#  if __has_builtin(__builtin_amdgcn_tensor_load_to_lds) && \
      __has_builtin(__builtin_amdgcn_s_wait_tensorcnt)
#    define USE_TDM 1
#  endif
#endif
#ifndef USE_TDM
#  define USE_TDM 0
#endif

__device__ __forceinline__ v8f wmma_f16(v16h a, v16h b, v8f c) {
  return __builtin_amdgcn_wmma_f32_16x16x32_f16(false, a, false, b, (short)0, c,
                                                false, false);
}

// A-fragment (16x32 f16) from LDS tile with row stride 32 halves.
// lanes 0-15: row=lane, K 0..15 ; lanes 16-31: row=lane-16, K 16..31
__device__ __forceinline__ v16h load_a_frag(const _Float16* tile, int lane) {
  const int row = lane & 15;
  const int hi8 = (lane >> 4) << 3;
  const _Float16* p = tile + row * 32;
  v8h lo = *(const v8h*)(p + 2 * hi8);
  v8h hi = *(const v8h*)(p + 8 + 2 * hi8);
  v16h r;
#pragma unroll
  for (int i = 0; i < 8; ++i) { r[i] = lo[i]; r[8 + i] = hi[i]; }
  return r;
}

#if USE_TDM
// Generic pointer to an LDS object: low 32 bits are the LDS byte offset
// (LDS aperture: ADDR[31:0] is the wave-relative LDS address).
__device__ __forceinline__ unsigned lds_off(const void* p) {
  return (unsigned)(unsigned long long)p;
}

// Issue one 2-D TDM tile load (Global -> LDS), f16 elements.
// tile_d0 = contiguous run (elements), tile_d1 = number of lines,
// stride_d0 = line stride in elements; tensor dims for OOB checks.
__device__ __forceinline__ void tdm_load_2d(unsigned lds_byte_off,
                                            const void* gptr,
                                            unsigned tile_d0, unsigned tile_d1,
                                            unsigned tensor_d0, unsigned tensor_d1,
                                            unsigned stride_d0) {
  unsigned long long ga = (unsigned long long)gptr;
  u32x4 g0;
  g0[0] = 1u;                                        // count=1, user descriptor
  g0[1] = lds_byte_off;                              // lds_addr
  g0[2] = (unsigned)(ga & 0xffffffffu);              // global_addr[31:0]
  g0[3] = (unsigned)((ga >> 32) & 0x01ffffffu)       // global_addr[56:32]
          | (2u << 30);                              // type=2 ("image")
  i32x8 g1;
  g1[0] = (int)(1u << 16);                           // data_size=1 (2 bytes)
  g1[1] = (int)((tensor_d0 & 0xffffu) << 16);        // tensor_dim0[15:0]
  g1[2] = (int)((tensor_d0 >> 16) | ((tensor_d1 & 0xffffu) << 16));
  g1[3] = (int)((tensor_d1 >> 16) | ((tile_d0 & 0xffffu) << 16));
  g1[4] = (int)(tile_d1 & 0xffffu);                  // tile_dim1 (tile_dim2=0)
  g1[5] = (int)stride_d0;                            // tensor_dim0_stride[31:0]
  g1[6] = 0;
  g1[7] = 0;
  i32x4 z4 = {0, 0, 0, 0};
#if __clang_major__ >= 23
  i32x8 z8 = {0, 0, 0, 0, 0, 0, 0, 0};
  __builtin_amdgcn_tensor_load_to_lds(g0, g1, z4, z4, z8, 0);
#else
  __builtin_amdgcn_tensor_load_to_lds(g0, g1, z4, z4, 0);
#endif
}
#endif  // USE_TDM

// ---------------------------------------------------------------------------
// Kernel 1: X(4096x768 f32) @ W(768x768 f32) + bias -> f16, head-split layout.
// mode 0: out[((b*12+h)*2048+s)*64+d]   (Q, K)
// mode 1: out[((b*12+h)*64+d)*2048+s]   (V transposed)
// ---------------------------------------------------------------------------
__global__ __launch_bounds__(256) void proj_kernel(
    const float* __restrict__ X, const float* __restrict__ W,
    const float* __restrict__ bias, _Float16* __restrict__ out,
    int mode, float scale)
{
  __shared__ _Float16 As[128 * 32];  // row-major, stride 32 halves
  __shared__ _Float16 Bs[64 * 32];   // column-major: col*32 + k

  const int t    = threadIdx.x;
  const int lane = t & 31;
  const int wave = t >> 5;
  const int blockRow = blockIdx.y * 128;
  const int blockCol = blockIdx.x * 64;
  const int waveRow  = (wave >> 1) * 32;
  const int waveCol  = (wave & 1) * 32;

  v8f acc[2][2];
#pragma unroll
  for (int a = 0; a < 2; ++a)
#pragma unroll
    for (int b = 0; b < 2; ++b)
#pragma unroll
      for (int i = 0; i < 8; ++i) acc[a][b][i] = 0.0f;

  for (int k0 = 0; k0 < DMODEL; k0 += 32) {
    __syncthreads();
    // stage A: 128x32, f32 -> f16, 4 float4 per thread
#pragma unroll
    for (int i = 0; i < 4; ++i) {
      int idx = t + i * 256;
      int r   = idx >> 3;
      int c4  = (idx & 7) << 2;
      const float* gp = X + (size_t)(blockRow + r) * DMODEL + k0 + c4;
      v4f x = *(const v4f*)gp;
      if (k0 + 32 < DMODEL) __builtin_prefetch(gp + 32, 0, 1);  // next K slab
      v4h h;
#pragma unroll
      for (int j = 0; j < 4; ++j) h[j] = (_Float16)x[j];
      *(v4h*)(As + r * 32 + c4) = h;
    }
    // stage B: 32x64 tile of W, transposed into column-major LDS
#pragma unroll
    for (int i = 0; i < 2; ++i) {
      int idx = t + i * 256;
      int kk  = idx >> 4;
      int c4  = (idx & 15) << 2;
      const float* gp = W + (size_t)(k0 + kk) * DMODEL + blockCol + c4;
      v4f x = *(const v4f*)gp;
      if (k0 + 32 < DMODEL) __builtin_prefetch(gp + (size_t)32 * DMODEL, 0, 1);
#pragma unroll
      for (int j = 0; j < 4; ++j) Bs[(c4 + j) * 32 + kk] = (_Float16)x[j];
    }
    __syncthreads();

    v16h a0 = load_a_frag(As + (waveRow +  0) * 32, lane);
    v16h a1 = load_a_frag(As + (waveRow + 16) * 32, lane);
    const int hi16 = (lane >> 4) << 4;
    const int nl   = lane & 15;
    v16h b0 = *(const v16h*)(Bs + (waveCol +      nl) * 32 + hi16);
    v16h b1 = *(const v16h*)(Bs + (waveCol + 16 + nl) * 32 + hi16);
    acc[0][0] = wmma_f16(a0, b0, acc[0][0]);
    acc[0][1] = wmma_f16(a0, b1, acc[0][1]);
    acc[1][0] = wmma_f16(a1, b0, acc[1][0]);
    acc[1][1] = wmma_f16(a1, b1, acc[1][1]);
  }

  const int hi = lane >> 4;
  const int nl = lane & 15;
#pragma unroll
  for (int rt = 0; rt < 2; ++rt)
#pragma unroll
    for (int ct = 0; ct < 2; ++ct)
#pragma unroll
      for (int i = 0; i < 8; ++i) {
        int grow = blockRow + waveRow + rt * 16 + hi * 8 + i;
        int gcol = blockCol + waveCol + ct * 16 + nl;
        float v = (acc[rt][ct][i] + bias[gcol]) * scale;
        int b = grow >> 11, s = grow & (SEQ - 1);
        int h = gcol >> 6,  d = gcol & (DEPTH - 1);
        size_t off = (mode == 0)
          ? ((((size_t)b * NUM_HEADS + h) * SEQ + s) * DEPTH + d)
          : ((((size_t)b * NUM_HEADS + h) * DEPTH + d) * SEQ + s);
        out[off] = (_Float16)v;
      }
}

// ---------------------------------------------------------------------------
// Kernel 2: flash attention, TDM double-buffered K/V staging.
// One block = 8 waves * 16 queries = 128 queries of one (b,h).
// ---------------------------------------------------------------------------
__global__ __launch_bounds__(256) void attn_kernel(
    const _Float16* __restrict__ Qh,  // (24, 2048, 64), pre-scaled by 1/8
    const _Float16* __restrict__ Kh,  // (24, 2048, 64)
    const _Float16* __restrict__ Vt,  // (24, 64, 2048)
    _Float16* __restrict__ ctx)       // (4096, 768)
{
  __shared__ _Float16 Ks[2][32 * 64];  // [key][d] — column n=key contiguous in d
  __shared__ _Float16 Vs[2][64 * 32];  // [d][kk]  — column n=d contiguous in kk
  __shared__ _Float16 Ps[8][16 * 32];

  const int t    = threadIdx.x;
  const int lane = t & 31;
  const int wave = t >> 5;
  const int bh    = blockIdx.y;
  const int qBase = blockIdx.x * 128 + wave * 16;
  const _Float16* Qbase = Qh + ((size_t)bh * SEQ + qBase) * DEPTH;
  const _Float16* Kbh   = Kh + (size_t)bh * SEQ * DEPTH;
  const _Float16* Vbh   = Vt + (size_t)bh * DEPTH * SEQ;

  const int row  = lane & 15;
  const int hi8  = (lane >> 4) << 3;
  const int hi16 = (lane >> 4) << 4;
  const int nl   = lane & 15;

  // Q A-fragments straight from global (rows are contiguous 64 halves)
  v16h qa[2];
#pragma unroll
  for (int ks = 0; ks < 2; ++ks) {
    const _Float16* p = Qbase + row * DEPTH + ks * 32;
    v8h lo = *(const v8h*)(p + 2 * hi8);
    v8h hi = *(const v8h*)(p + 8 + 2 * hi8);
#pragma unroll
    for (int i = 0; i < 8; ++i) { qa[ks][i] = lo[i]; qa[ks][8 + i] = hi[i]; }
  }

  float m[8], l[8];
  v8f o[4];
#pragma unroll
  for (int i = 0; i < 8; ++i) { m[i] = -1e30f; l[i] = 0.0f; }
#pragma unroll
  for (int n = 0; n < 4; ++n)
#pragma unroll
    for (int i = 0; i < 8; ++i) o[n][i] = 0.0f;

  const int NBLK = SEQ / 32;  // 64

#if USE_TDM
  if (wave == 0) {
    // prime the pipeline: block 0 into buffer 0
    tdm_load_2d(lds_off(&Ks[0][0]), Kbh, /*tile*/64, 32, /*tensor*/64, SEQ, 64);
    tdm_load_2d(lds_off(&Vs[0][0]), Vbh, /*tile*/32, 64, /*tensor*/SEQ, DEPTH, SEQ);
  }
#endif

  for (int i = 0; i < NBLK; ++i) {
    const int buf = i & 1;
    const int kb  = i * 32;

#if USE_TDM
    if (wave == 0) {
      if (i + 1 < NBLK) {
        // issue block i+1 into the other buffer; its previous readers
        // retired at the barrier closing iteration i-1.
        tdm_load_2d(lds_off(&Ks[buf ^ 1][0]),
                    Kbh + (size_t)(i + 1) * 32 * DEPTH, 64, 32, 64, SEQ, 64);
        tdm_load_2d(lds_off(&Vs[buf ^ 1][0]),
                    Vbh + (size_t)(i + 1) * 32, 32, 64, SEQ, DEPTH, SEQ);
        __builtin_amdgcn_s_wait_tensorcnt(2);  // block i complete; i+1 in flight
      } else {
        __builtin_amdgcn_s_wait_tensorcnt(0);
      }
    }
    __syncthreads();  // publish TDM-written LDS to all waves
#else
    __syncthreads();
    {  // stage K block: 32 keys x 64 d (one 16B copy per thread)
      int key = t >> 3, seg = (t & 7) << 3;
      *(v8h*)(&Ks[buf][0] + key * 64 + seg) =
          *(const v8h*)(Kbh + (size_t)(kb + key) * DEPTH + seg);
    }
    {  // stage V block: 64 d x 32 keys from Vt rows
      int d = t >> 2, seg = (t & 3) << 3;
      *(v8h*)(&Vs[buf][0] + d * 32 + seg) =
          *(const v8h*)(Vbh + (size_t)d * SEQ + kb + seg);
    }
    __syncthreads();
#endif

    const _Float16* Kbuf = &Ks[buf][0];
    const _Float16* Vbuf = &Vs[buf][0];

    // S = Q K^T  (16q x 32k), f32 accum
    v8f s0, s1;
#pragma unroll
    for (int j = 0; j < 8; ++j) { s0[j] = 0.0f; s1[j] = 0.0f; }
#pragma unroll
    for (int ks = 0; ks < 2; ++ks) {
      v16h b0 = *(const v16h*)(Kbuf + (nl)      * 64 + ks * 32 + hi16);
      v16h b1 = *(const v16h*)(Kbuf + (16 + nl) * 64 + ks * 32 + hi16);
      s0 = wmma_f16(qa[ks], b0, s0);
      s1 = wmma_f16(qa[ks], b1, s1);
    }

    // online softmax: row r = (lane>=16)*8 + j lives across a 16-lane half
    float alpha[8];
#pragma unroll
    for (int j = 0; j < 8; ++j) {
      float v = fmaxf(s0[j], s1[j]);
#pragma unroll
      for (int off = 1; off < 16; off <<= 1) v = fmaxf(v, __shfl_xor(v, off, 32));
      float mn = fmaxf(m[j], v);
      alpha[j] = __expf(m[j] - mn);
      m[j] = mn;
      float p0 = __expf(s0[j] - mn);
      float p1 = __expf(s1[j] - mn);
      s0[j] = p0; s1[j] = p1;
      float rs = p0 + p1;
#pragma unroll
      for (int off = 1; off < 16; off <<= 1) rs += __shfl_xor(rs, off, 32);
      l[j] = l[j] * alpha[j] + rs;
    }
#pragma unroll
    for (int n = 0; n < 4; ++n)
#pragma unroll
      for (int j = 0; j < 8; ++j) o[n][j] *= alpha[j];

    // repack P: C-layout regs -> LDS -> A-layout fragment (wave-local)
    _Float16* Pw = Ps[wave];
#pragma unroll
    for (int j = 0; j < 8; ++j) {
      int pr = hi8 + j;
      Pw[pr * 32 + nl]      = (_Float16)s0[j];
      Pw[pr * 32 + 16 + nl] = (_Float16)s1[j];
    }
    v16h pa;
    {
      const _Float16* p = Pw + row * 32;
      v8h lo = *(const v8h*)(p + 2 * hi8);
      v8h hi = *(const v8h*)(p + 8 + 2 * hi8);
#pragma unroll
      for (int j = 0; j < 8; ++j) { pa[j] = lo[j]; pa[8 + j] = hi[j]; }
    }

    // O += P * V  (N = 4 tiles of d)
#pragma unroll
    for (int n = 0; n < 4; ++n) {
      v16h bv = *(const v16h*)(Vbuf + (n * 16 + nl) * 32 + hi16);
      o[n] = wmma_f16(pa, bv, o[n]);
    }

    __syncthreads();  // all reads of this buffer retired before TDM reuses it
  }

  // epilogue: O /= l, store ctx (b,s,h*64+d) as f16
  const int h = bh % NUM_HEADS;
  const int b = bh / NUM_HEADS;
#pragma unroll
  for (int n = 0; n < 4; ++n)
#pragma unroll
    for (int j = 0; j < 8; ++j) {
      int q = qBase + hi8 + j;
      int d = n * 16 + nl;
      float val = o[n][j] / l[j];
      ctx[((size_t)b * SEQ + q) * DMODEL + h * DEPTH + d] = (_Float16)val;
    }
}

// ---------------------------------------------------------------------------
// Kernel 3: ctx(4096x768 f16) @ Wo(768x768 f32) + bo -> f32 out
// ---------------------------------------------------------------------------
__global__ __launch_bounds__(256) void oproj_kernel(
    const _Float16* __restrict__ X, const float* __restrict__ W,
    const float* __restrict__ bias, float* __restrict__ out)
{
  __shared__ _Float16 As[128 * 32];
  __shared__ _Float16 Bs[64 * 32];

  const int t    = threadIdx.x;
  const int lane = t & 31;
  const int wave = t >> 5;
  const int blockRow = blockIdx.y * 128;
  const int blockCol = blockIdx.x * 64;
  const int waveRow  = (wave >> 1) * 32;
  const int waveCol  = (wave & 1) * 32;

  v8f acc[2][2];
#pragma unroll
  for (int a = 0; a < 2; ++a)
#pragma unroll
    for (int b = 0; b < 2; ++b)
#pragma unroll
      for (int i = 0; i < 8; ++i) acc[a][b][i] = 0.0f;

  for (int k0 = 0; k0 < DMODEL; k0 += 32) {
    __syncthreads();
    // stage A: f16 copy, 2 x 16B per thread
#pragma unroll
    for (int i = 0; i < 2; ++i) {
      int idx = t + i * 256;
      int r   = idx >> 2;
      int seg = (idx & 3) << 3;
      const _Float16* gp = X + (size_t)(blockRow + r) * DMODEL + k0 + seg;
      if (k0 + 32 < DMODEL) __builtin_prefetch(gp + 32, 0, 1);
      *(v8h*)(As + r * 32 + seg) = *(const v8h*)gp;
    }
    // stage B transposed
#pragma unroll
    for (int i = 0; i < 2; ++i) {
      int idx = t + i * 256;
      int kk  = idx >> 4;
      int c4  = (idx & 15) << 2;
      const float* gp = W + (size_t)(k0 + kk) * DMODEL + blockCol + c4;
      v4f x = *(const v4f*)gp;
      if (k0 + 32 < DMODEL) __builtin_prefetch(gp + (size_t)32 * DMODEL, 0, 1);
#pragma unroll
      for (int j = 0; j < 4; ++j) Bs[(c4 + j) * 32 + kk] = (_Float16)x[j];
    }
    __syncthreads();

    v16h a0 = load_a_frag(As + (waveRow +  0) * 32, lane);
    v16h a1 = load_a_frag(As + (waveRow + 16) * 32, lane);
    const int hi16 = (lane >> 4) << 4;
    const int nl   = lane & 15;
    v16h b0 = *(const v16h*)(Bs + (waveCol +      nl) * 32 + hi16);
    v16h b1 = *(const v16h*)(Bs + (waveCol + 16 + nl) * 32 + hi16);
    acc[0][0] = wmma_f16(a0, b0, acc[0][0]);
    acc[0][1] = wmma_f16(a0, b1, acc[0][1]);
    acc[1][0] = wmma_f16(a1, b0, acc[1][0]);
    acc[1][1] = wmma_f16(a1, b1, acc[1][1]);
  }

  const int hi = lane >> 4;
  const int nl = lane & 15;
#pragma unroll
  for (int rt = 0; rt < 2; ++rt)
#pragma unroll
    for (int ct = 0; ct < 2; ++ct)
#pragma unroll
      for (int i = 0; i < 8; ++i) {
        int grow = blockRow + waveRow + rt * 16 + hi * 8 + i;
        int gcol = blockCol + waveCol + ct * 16 + nl;
        out[(size_t)grow * DMODEL + gcol] = acc[rt][ct][i] + bias[gcol];
      }
}

// ---------------------------------------------------------------------------
extern "C" void kernel_launch(void* const* d_in, const int* in_sizes, int n_in,
                              void* d_out, int out_size, void* d_ws, size_t ws_size,
                              hipStream_t stream) {
  const float* v  = (const float*)d_in[0];
  const float* k  = (const float*)d_in[1];
  const float* q  = (const float*)d_in[2];
  const float* wq = (const float*)d_in[3];
  const float* bq = (const float*)d_in[4];
  const float* wk = (const float*)d_in[5];
  const float* bk = (const float*)d_in[6];
  const float* wv = (const float*)d_in[7];
  const float* bv = (const float*)d_in[8];
  const float* wo = (const float*)d_in[9];
  const float* bo = (const float*)d_in[10];

  const size_t ELEMS = (size_t)BH * SEQ * DEPTH;  // 3,145,728
  _Float16* Qh = (_Float16*)d_ws;
  _Float16* Kh = Qh + ELEMS;
  _Float16* Vt = Kh + ELEMS;
  _Float16* CT = Vt + ELEMS;

  dim3 gridP(DMODEL / 64, ROWS / 128);  // (12, 32)
  proj_kernel<<<gridP, 256, 0, stream>>>(q, wq, bq, Qh, 0, 0.125f);
  proj_kernel<<<gridP, 256, 0, stream>>>(k, wk, bk, Kh, 0, 1.0f);
  proj_kernel<<<gridP, 256, 0, stream>>>(v, wv, bv, Vt, 1, 1.0f);

  attn_kernel<<<dim3(SEQ / 128, BH), 256, 0, stream>>>(Qh, Kh, Vt, CT);

  oproj_kernel<<<gridP, 256, 0, stream>>>(CT, wo, bo, (float*)d_out);
}
